// AttentionOut_63711544869147
// MI455X (gfx1250) — matrix operand order repaired and checked by reading, not compiled, not run on hardware
//
#include <hip/hip_runtime.h>

typedef __attribute__((ext_vector_type(16))) __bf16 v16bf;
typedef __attribute__((ext_vector_type(8)))  float  v8f;

#define N_HEADS   16
#define D_HEAD    64
#define D_MODEL   1024
#define SEQ       2048
#define INV_SCALE 0.125f          // 1/sqrt(64)
#define NEG_BIG   (-1.0e30f)

#define WMMA_BF16(a, b, c) \
    __builtin_amdgcn_wmma_f32_16x16x32_bf16(false, (a), false, (b), (short)0, (c), false, false)

// ---------------------------------------------------------------------------
// Kernel 1: flash attention. One workgroup = 4 waves = (b, h, 64 query rows).
// Each wave owns a 16-row Q tile; loop over 32-key chunks staged in LDS.
// z[b, s, h*64+d] written to workspace (fp32).
// ---------------------------------------------------------------------------
__global__ __launch_bounds__(128) void attn_fwd(const float* __restrict__ q,
                                                const float* __restrict__ k,
                                                const float* __restrict__ v,
                                                float* __restrict__ z) {
    __shared__ __bf16 Kbf[32 * 64];       // K chunk, row-major [key][d]
    __shared__ __bf16 Vt[64 * 32];        // V chunk, transposed [d][key]
    __shared__ __bf16 Pb[4][16 * 32];     // per-wave P tile [m][key]

    const int tid  = threadIdx.x;
    const int wave = tid >> 5;
    const int lane = tid & 31;
    const int n    = lane & 15;           // column index (C/B layout)
    const int hi   = lane >> 4;           // half-wave select

    const int nQB = SEQ / 64;
    const int bid = blockIdx.x;
    const int b   = bid / (N_HEADS * nQB);
    const int rem = bid % (N_HEADS * nQB);
    const int h   = rem / nQB;
    const int qb  = rem % nQB;
    const int q0  = qb * 64 + wave * 16;  // this wave's first query row

    const float* qbase = q + ((size_t)b * SEQ) * D_MODEL + h * D_HEAD;
    const float* kbase = k + ((size_t)b * SEQ) * D_MODEL + h * D_HEAD;
    const float* vbase = v + ((size_t)b * SEQ) * D_MODEL + h * D_HEAD;

    // ---- Load Q tile into A-matrix layout (two K=32 chunks along d) ----
    // A operand (16x32), k order chosen here (B matches): lane m (&15),
    // element i -> K = 16*(lane>=16) + i
    v16bf aQ[2];
    {
        const float* qrow = qbase + (size_t)(q0 + n) * D_MODEL;
        const int kb = hi * 8;
#pragma unroll
        for (int c2 = 0; c2 < 2; ++c2) {
#pragma unroll
            for (int i = 0; i < 16; ++i) {
                const int kk = hi * 16 + i;
                aQ[c2][i] = (__bf16)qrow[c2 * 32 + kk];
            }
        }
    }

    const v8f vzero = {0.f, 0.f, 0.f, 0.f, 0.f, 0.f, 0.f, 0.f};
    v8f   O[4] = {vzero, vzero, vzero, vzero};  // 16x64 output accumulator
    float mprev[8], lsum[8];
#pragma unroll
    for (int r = 0; r < 8; ++r) { mprev[r] = NEG_BIG; lsum[r] = 0.f; }

    const int nchunks = qb * 2 + 2;  // WG-uniform: covers keys [0, qb*64+64)
    for (int c = 0; c < nchunks; ++c) {
        const int key0 = c * 32;
        __syncthreads();  // protect LDS from previous iteration's readers
        // ---- cooperative stage: K chunk (row major) and V chunk (transposed)
        for (int e = tid; e < 32 * 64; e += 128) {
            const int kk = e >> 6;   // key within chunk
            const int dd = e & 63;   // d
            Kbf[kk * 64 + dd] = (__bf16)kbase[(size_t)(key0 + kk) * D_MODEL + dd];
            Vt[dd * 32 + kk]  = (__bf16)vbase[(size_t)(key0 + kk) * D_MODEL + dd];
        }
        __syncthreads();

        // ---- S = Q @ K^T  (two 16x16 key sub-tiles, K-dim = d split 2x32)
        v8f Sv[2];
#pragma unroll
        for (int t = 0; t < 2; ++t) {
            v8f cacc = vzero;
#pragma unroll
            for (int dc = 0; dc < 2; ++dc) {
                // B operand (32x16), same k order as aQ: lane col n, element i -> row hi*16+i
                v16bf bK;
                const int dofs = dc * 32 + hi * 16;
#pragma unroll
                for (int i = 0; i < 16; ++i)
                    bK[i] = Kbf[(t * 16 + n) * 64 + dofs + i];
                cacc = WMMA_BF16(aQ[dc], bK, cacc);
            }
            Sv[t] = cacc;
        }

        // ---- online softmax (C layout: lane holds rows r+8*hi at col n) ----
        float alpha[8];
#pragma unroll
        for (int r = 0; r < 8; ++r) {
            const int mg = q0 + r + 8 * hi;
            float s0 = Sv[0][r] * INV_SCALE;
            float s1 = Sv[1][r] * INV_SCALE;
            s0 = ((key0 + n)      <= mg) ? s0 : NEG_BIG;
            s1 = ((key0 + 16 + n) <= mg) ? s1 : NEG_BIG;
            float mx = fmaxf(s0, s1);
            mx = fmaxf(mx, __shfl_xor(mx, 1, 32));
            mx = fmaxf(mx, __shfl_xor(mx, 2, 32));
            mx = fmaxf(mx, __shfl_xor(mx, 4, 32));
            mx = fmaxf(mx, __shfl_xor(mx, 8, 32));
            const float mn = fmaxf(mprev[r], mx);
            const float al = __expf(mprev[r] - mn);
            const float p0 = __expf(s0 - mn);
            const float p1 = __expf(s1 - mn);
            float ps = p0 + p1;
            ps += __shfl_xor(ps, 1, 32);
            ps += __shfl_xor(ps, 2, 32);
            ps += __shfl_xor(ps, 4, 32);
            ps += __shfl_xor(ps, 8, 32);
            lsum[r]  = lsum[r] * al + ps;
            mprev[r] = mn;
            alpha[r] = al;
            Sv[0][r] = p0;
            Sv[1][r] = p1;
        }
#pragma unroll
        for (int dt = 0; dt < 4; ++dt)
#pragma unroll
            for (int r = 0; r < 8; ++r) O[dt][r] *= alpha[r];

        // ---- re-layout P: C format -> LDS -> A format (cross-lane) ----
        __bf16* pw = &Pb[wave][0];
#pragma unroll
        for (int r = 0; r < 8; ++r) {
            const int m = r + 8 * hi;
            pw[m * 32 + n]      = (__bf16)Sv[0][r];
            pw[m * 32 + 16 + n] = (__bf16)Sv[1][r];
        }
        // DS ops are in-order within a wave; fence compiler + wait for stores
        asm volatile("s_wait_dscnt 0x0" ::: "memory");

        v16bf aP;
        {
            const int kb = hi * 8;
#pragma unroll
            for (int i = 0; i < 16; ++i) {
                const int kk = hi * 16 + i;
                aP[i] = pw[n * 32 + kk];
            }
        }
        // ---- O += P @ V  (4 d-tiles of 16, K-dim = 32 keys) ----
#pragma unroll
        for (int dt = 0; dt < 4; ++dt) {
            v16bf bV;
#pragma unroll
            for (int i = 0; i < 16; ++i)
                bV[i] = Vt[(dt * 16 + n) * 32 + hi * 16 + i];
            O[dt] = WMMA_BF16(aP, bV, O[dt]);
        }
    }

    // ---- normalize and store z ----
    float* zb = z + ((size_t)(b * SEQ + q0)) * D_MODEL + h * D_HEAD;
#pragma unroll
    for (int dt = 0; dt < 4; ++dt) {
#pragma unroll
        for (int r = 0; r < 8; ++r) {
            const int m = r + 8 * hi;
            zb[(size_t)m * D_MODEL + dt * 16 + n] = O[dt][r] / lsum[r];
        }
    }
}

// ---------------------------------------------------------------------------
// Kernel 2: out = Z @ W_O + b_O.  Z:(B*S,1024) hd-major, W_O:(1024,1024).
// WG = 4 waves = 16 rows x 64 cols; A tile shared via LDS, 1 WMMA per chunk.
// ---------------------------------------------------------------------------
__global__ __launch_bounds__(128) void proj_fwd(const float* __restrict__ Z,
                                                const float* __restrict__ W,
                                                const float* __restrict__ bias,
                                                float* __restrict__ out) {
    __shared__ __bf16 Abf[16 * 32];
    const int tid  = threadIdx.x;
    const int wave = tid >> 5;
    const int lane = tid & 31;
    const int n    = lane & 15;
    const int hi   = lane >> 4;

    const int nColB = D_MODEL / 64;                  // 16 col-blocks
    const int row0  = (blockIdx.x / nColB) * 16;
    const int col0  = (blockIdx.x % nColB) * 64 + wave * 16;

    const v8f vzero = {0.f, 0.f, 0.f, 0.f, 0.f, 0.f, 0.f, 0.f};
    v8f acc = vzero;

    for (int c = 0; c < D_MODEL / 32; ++c) {
        const int k0 = c * 32;
        __syncthreads();
        for (int e = tid; e < 16 * 32; e += 128) {
            const int m  = e >> 5;
            const int kk = e & 31;
            Abf[m * 32 + kk] = (__bf16)Z[(size_t)(row0 + m) * D_MODEL + k0 + kk];
        }
        __syncthreads();

        v16bf aA;
        {
            const int kb = hi * 8;
#pragma unroll
            for (int i = 0; i < 16; ++i) {
                const int kk = hi * 16 + i;
                aA[i] = Abf[n * 32 + kk];
            }
        }
        v16bf bW;
#pragma unroll
        for (int i = 0; i < 16; ++i)
            bW[i] = (__bf16)W[(size_t)(k0 + hi * 16 + i) * D_MODEL + col0 + n];

        acc = WMMA_BF16(aA, bW, acc);
    }

    const float bc = bias[col0 + n];
#pragma unroll
    for (int r = 0; r < 8; ++r) {
        const int m = r + 8 * hi;
        out[(size_t)(row0 + m) * D_MODEL + col0 + n] = acc[r] + bc;
    }
}

// ---------------------------------------------------------------------------
// Kernel 3: residual passthrough (first output of the tuple).
// ---------------------------------------------------------------------------
__global__ void copy_res(const float4* __restrict__ in, float4* __restrict__ out,
                         int nvec) {
    const int i = blockIdx.x * blockDim.x + threadIdx.x;
    if (i < nvec) out[i] = in[i];
}

extern "C" void kernel_launch(void* const* d_in, const int* in_sizes, int n_in,
                              void* d_out, int out_size, void* d_ws, size_t ws_size,
                              hipStream_t stream) {
    const float* residual = (const float*)d_in[0];
    const float* q        = (const float*)d_in[1];
    const float* k        = (const float*)d_in[2];
    const float* v        = (const float*)d_in[3];
    const float* W_O      = (const float*)d_in[4];
    const float* b_O      = (const float*)d_in[5];
    float* out = (float*)d_out;
    float* zws = (float*)d_ws;  // B*S*1024 fp32 = 16 MB scratch for z

    const int B   = in_sizes[0] / (SEQ * D_MODEL);  // batch from residual size
    const int BSM = B * SEQ * D_MODEL;

    // 1) residual copy -> first half of d_out
    const int nvec = BSM / 4;
    copy_res<<<(nvec + 255) / 256, 256, 0, stream>>>((const float4*)residual,
                                                     (float4*)out, nvec);
    // 2) flash attention -> z in workspace
    attn_fwd<<<B * N_HEADS * (SEQ / 64), 128, 0, stream>>>(q, k, v, zws);
    // 3) output projection -> second half of d_out
    proj_fwd<<<(B * SEQ / 16) * (D_MODEL / 64), 128, 0, stream>>>(zws, W_O, b_O,
                                                                  out + BSM);
}
